// LFQQuantizerEMA_21895743275556
// MI455X (gfx1250) — compile-verified, matrix operand order checked by reading, not run on hardware
//
#include <hip/hip_runtime.h>
#include <hip/hip_bf16.h>

// ---------------------------------------------------------------------------
// VQ-VAE EMA quantizer for MI455X (gfx1250, wave32).
// Phase 1 (dominant): fused  argmin_k ( c_sq[k] - 2 * z_e . codebook[k] )
//   via bf16x3 fp32-emulated WMMA (v_wmma_f32_16x16x32_bf16), with the
//   most-reused operand (codebook hi) staged into LDS by the Tensor Data
//   Mover (tensor_load_to_lds, TENSORcnt), double-buffered per 64-code strip.
// Phase 2: gather z_q, scatter-add EMA sums with global f32 atomics, finalize.
// ---------------------------------------------------------------------------

#define B_N 32768
#define K_N 8192
#define D_N 512

#define STRIP  64                  // codes per TDM strip
#define NT     (STRIP / 16)        // 4 accumulator tiles per wave
#define NSTRIP (K_N / STRIP)       // 128 strips
#define STRIP_BYTES (STRIP * D_N * 2)   // 64 KB of bf16 per buffer

#define DECAYF 0.99f
#define OMDF   (1.0f - DECAYF)
#define EPSF   1e-5f

typedef __attribute__((ext_vector_type(16))) __bf16 v16bf;
typedef __attribute__((ext_vector_type(8)))  float  v8f;
typedef __attribute__((ext_vector_type(4)))  unsigned int u32x4;
typedef __attribute__((ext_vector_type(8)))  int   i32x8;
typedef __attribute__((ext_vector_type(4)))  int   i32x4;

// Load one 16x32 bf16 WMMA operand fragment from a row-major matrix
// (global or LDS).  Lane layout (ISA 7.12.2, 16-bit A-matrix 16x32):
// lane l%16 holds row l%16; lanes 0-15 hold K=[0..7,16..23], lanes 16-31
// hold K=[8..15,24..31]  => two 16-byte loads per lane.
template <typename P>
__device__ __forceinline__ v16bf load_frag16x32(P rowptr, int hf) {
  union { v16bf v; uint4 q[2]; } f;
  f.q[0] = *(const uint4*)(rowptr + hf * 8);
  f.q[1] = *(const uint4*)(rowptr + 16 + hf * 8);
  return f.v;
}

// Issue one TDM copy: STRIP contiguous bf16 codebook rows -> LDS.
// D# group0: count=1, lds_addr, 57-bit global_addr, type=2 ("image").
// D# group1: data_size=4B, tensor_dim0=stride0=tile_dim0=256 dwords/row,
//            tensor_dim1=K, tile_dim1=STRIP  (fully contiguous transfer).
__device__ __forceinline__ void tdm_load_strip(const __bf16* gsrc,
                                               unsigned lds_byte_off) {
  unsigned long long ga = (unsigned long long)gsrc;
  u32x4 g0;
  g0[0] = 1u;                                   // count=1, user mode
  g0[1] = lds_byte_off;                         // lds_addr
  g0[2] = (unsigned)ga;                         // global_addr[31:0]
  g0[3] = (unsigned)(ga >> 32) | (2u << 30);    // global_addr[56:32] | type=2

  const unsigned td0   = (D_N * 2) / 4;         // 256 dwords per bf16 row
  const unsigned td1   = K_N;
  const unsigned tile0 = td0;
  const unsigned tile1 = STRIP;
  const unsigned str0  = td0;
  i32x8 g1;
  g1[0] = (int)(2u << 16);                                  // data_size = 4B
  g1[1] = (int)((td0 & 0xFFFFu) << 16);                     // tensor_dim0 lo
  g1[2] = (int)((td0 >> 16) | ((td1 & 0xFFFFu) << 16));     // dim0 hi | dim1 lo
  g1[3] = (int)((td1 >> 16) | (tile0 << 16));               // dim1 hi | tile0
  g1[4] = (int)tile1;                                       // tile1 (tile2=0)
  g1[5] = (int)str0;                                        // dim0_stride lo
  g1[6] = 0;                                                // stride0 hi | stride1 lo
  g1[7] = 0;                                                // stride1 hi
  i32x4 z4 = {0, 0, 0, 0};
#if __clang_major__ >= 23
  i32x8 z8 = {0, 0, 0, 0, 0, 0, 0, 0};
  __builtin_amdgcn_tensor_load_to_lds(g0, g1, z4, z4, z8, 0);
#else
  __builtin_amdgcn_tensor_load_to_lds(g0, g1, z4, z4, 0);
#endif
}

// ---- prep: split z_e into bf16 hi/lo ---------------------------------------
__global__ void k_convert_z(const float* __restrict__ z,
                            __bf16* __restrict__ zhi, __bf16* __restrict__ zlo) {
  size_t i = (size_t)blockIdx.x * blockDim.x + threadIdx.x;
  float x = z[i];
  __bf16 h = (__bf16)x;
  zhi[i] = h;
  zlo[i] = (__bf16)(x - (float)h);
}

// ---- prep: split codebook into bf16 hi/lo + per-code squared norm ----------
__global__ void k_convert_cb(const float* __restrict__ cb,
                             __bf16* __restrict__ chi, __bf16* __restrict__ clo,
                             float* __restrict__ csq) {
  int lane = threadIdx.x & 31;
  int wv   = threadIdx.x >> 5;
  int row  = blockIdx.x * 8 + wv;              // 8 waves/block, 1 row/wave
  const float* r = cb + (size_t)row * D_N;
  __bf16* hi = chi + (size_t)row * D_N;
  __bf16* lo = clo + (size_t)row * D_N;
  float s = 0.0f;
  for (int c = lane; c < D_N; c += 32) {
    float x = r[c];
    __bf16 h = (__bf16)x;
    hi[c] = h;
    lo[c] = (__bf16)(x - (float)h);
    s += x * x;
  }
#pragma unroll
  for (int m = 16; m; m >>= 1) s += __shfl_xor(s, m, 32);
  if (lane == 0) csq[row] = s;
}

// ---- prep: new_ema_codebook := DECAY * ema_codebook ; counts := 0 ----------
__global__ void k_init(const float* __restrict__ ecb,
                       float* __restrict__ necb_out, float* __restrict__ counts) {
  size_t i = (size_t)blockIdx.x * blockDim.x + threadIdx.x;
  necb_out[i] = DECAYF * ecb[i];
  if (i < K_N) counts[i] = 0.0f;
}

// ---- phase 1: fused bf16x3 WMMA distance + argmin --------------------------
// Block = 256 threads = 8 waves; each wave owns 16 z-rows.  Codebook hi is
// DMA'd strip-by-strip into double-buffered LDS by the TDM (wave 0 issues,
// s_wait_tensorcnt + barrier release the consumers); codebook lo and the
// z fragments stream through the global path.
__global__ void __launch_bounds__(256)
k_argmin(const __bf16* __restrict__ zhi, const __bf16* __restrict__ zlo,
         const __bf16* __restrict__ chi, const __bf16* __restrict__ clo,
         const float* __restrict__ csq,
         int* __restrict__ idx_ws, float* __restrict__ idx_out) {
  extern __shared__ __bf16 lds_b[];   // 2 * STRIP * D_N bf16 (128 KB)

  const int lane = threadIdx.x & 31;
  const int wv   = threadIdx.x >> 5;
  const int hf   = lane >> 4;          // lane half selects K-subrange / row+8
  const int lm   = lane & 15;          // row (A) / code (B) within tile
  const int m0   = blockIdx.x * 128 + wv * 16;

  const __bf16* arow_hi = zhi + (size_t)(m0 + lm) * D_N;
  const __bf16* arow_lo = zlo + (size_t)(m0 + lm) * D_N;

  float bestVal[8];
  int   bestIdx[8];
#pragma unroll
  for (int r = 0; r < 8; ++r) { bestVal[r] = 3.4e38f; bestIdx[r] = 0; }

  if (wv == 0) tdm_load_strip(chi, 0);          // prime buffer 0

  for (int s = 0; s < NSTRIP; ++s) {
    const int kc = s * STRIP;
    if (wv == 0) {
      if (s + 1 < NSTRIP) {
        tdm_load_strip(chi + (size_t)(kc + STRIP) * D_N,
                       (unsigned)(((s + 1) & 1) * STRIP_BYTES));
        __builtin_amdgcn_s_wait_tensorcnt(1);   // strip s done, s+1 in flight
      } else {
        __builtin_amdgcn_s_wait_tensorcnt(0);
      }
    }
    __syncthreads();                            // LDS strip s visible to all

    const __bf16* bstrip = lds_b + (size_t)(s & 1) * (STRIP * D_N);

    v8f acc[NT];
#pragma unroll
    for (int t = 0; t < NT; ++t)
#pragma unroll
      for (int e = 0; e < 8; ++e) acc[t][e] = 0.0f;

    if (kc + STRIP < K_N)   // keep the lo-strip warm in L2/L1
      __builtin_prefetch(clo + (size_t)(kc + STRIP + lm) * D_N, 0, 1);

#pragma unroll 2
    for (int dc = 0; dc < D_N; dc += 32) {
      v16bf ahi = load_frag16x32(arow_hi + dc, hf);
      v16bf alo = load_frag16x32(arow_lo + dc, hf);
#pragma unroll
      for (int t = 0; t < NT; ++t) {
        const int bl = 16 * t + lm;                              // local code
        v16bf bhi = load_frag16x32(bstrip + (size_t)bl * D_N + dc, hf);
        v16bf blo = load_frag16x32(clo + (size_t)(kc + bl) * D_N + dc, hf);
        // fp32-emulated dot: hi*hi + hi*lo + lo*hi
        acc[t] = __builtin_amdgcn_wmma_f32_16x16x32_bf16(
            false, ahi, false, bhi, (short)0, acc[t], false, false);
        acc[t] = __builtin_amdgcn_wmma_f32_16x16x32_bf16(
            false, ahi, false, blo, (short)0, acc[t], false, false);
        acc[t] = __builtin_amdgcn_wmma_f32_16x16x32_bf16(
            false, alo, false, bhi, (short)0, acc[t], false, false);
      }
    }

    // score = c_sq - 2*dot ; track per-slot running argmin
#pragma unroll
    for (int t = 0; t < NT; ++t) {
      int code = kc + 16 * t + lm;
      float cs = csq[code];
#pragma unroll
      for (int r = 0; r < 8; ++r) {
        float score = cs - 2.0f * acc[t][r];
        if (score < bestVal[r]) { bestVal[r] = score; bestIdx[r] = code; }
      }
    }
    __syncthreads();   // all waves done with buf (s&1) before TDM overwrites
  }

  // reduce across the 16 lanes sharing each row (xor masks keep lane-half)
#pragma unroll
  for (int r = 0; r < 8; ++r) {
    float v  = bestVal[r];
    int   ix = bestIdx[r];
#pragma unroll
    for (int m = 1; m < 16; m <<= 1) {
      float ov = __shfl_xor(v, m, 32);
      int   oi = __shfl_xor(ix, m, 32);
      if (ov < v || (ov == v && oi < ix)) { v = ov; ix = oi; }
    }
    if (lm == 0) {
      int row = m0 + r + 8 * hf;   // lanes 0..15 -> rows r ; 16..31 -> r+8
      idx_ws[row]  = ix;
      idx_out[row] = (float)ix;
    }
  }
}

// ---- phase 2: gather z_q + scatter EMA sums --------------------------------
__global__ void k_scatter(const float* __restrict__ z, const float* __restrict__ cb,
                          const int* __restrict__ idxw,
                          float* __restrict__ zq_out, float* __restrict__ necb_out,
                          float* __restrict__ counts) {
  size_t i = (size_t)blockIdx.x * blockDim.x + threadIdx.x;  // b*D + d
  int b = (int)(i >> 9);
  int d = (int)(i & (D_N - 1));
  int k = idxw[b];
  size_t src = (size_t)k * D_N + d;
  zq_out[i] = cb[src];
  atomicAdd(&necb_out[src], OMDF * z[i]);
  if (d == 0) atomicAdd(&counts[k], 1.0f);
}

// ---- finalize --------------------------------------------------------------
__global__ void k_cluster(const float* __restrict__ ecs, const float* __restrict__ counts,
                          float* __restrict__ ncs_out) {
  int k = blockIdx.x * blockDim.x + threadIdx.x;
  ncs_out[k] = DECAYF * ecs[k] + OMDF * counts[k];
}

__global__ void k_newcb(const float* __restrict__ necb, const float* __restrict__ ecs,
                        const float* __restrict__ counts, float* __restrict__ ncb_out) {
  size_t i = (size_t)blockIdx.x * blockDim.x + threadIdx.x;
  int k = (int)(i >> 9);
  float denom = DECAYF * ecs[k] + OMDF * counts[k] + EPSF;
  ncb_out[i] = necb[i] / denom;
}

// ---------------------------------------------------------------------------
extern "C" void kernel_launch(void* const* d_in, const int* in_sizes, int n_in,
                              void* d_out, int out_size, void* d_ws, size_t ws_size,
                              hipStream_t stream) {
  const float* z_e = (const float*)d_in[0];   // (B, D)
  const float* cb  = (const float*)d_in[1];   // (K, D)
  const float* ecs = (const float*)d_in[2];   // (K,)
  const float* ecb = (const float*)d_in[3];   // (K, D)

  // outputs concatenated in return order (all float)
  float* out   = (float*)d_out;
  float* zq    = out;                                   // B*D
  float* idx_f = zq + (size_t)B_N * D_N;                // B
  float* ncb   = idx_f + B_N;                           // K*D
  float* ncs   = ncb + (size_t)K_N * D_N;               // K
  float* necb  = ncs + K_N;                             // K*D (new_ema_codebook)

  // workspace layout
  char* ws = (char*)d_ws;
  __bf16* zhi   = (__bf16*)(ws);                                 // 32 MB
  __bf16* zlo   = (__bf16*)(ws + (size_t)B_N * D_N * 2);         // 32 MB
  __bf16* chi   = (__bf16*)(ws + (size_t)B_N * D_N * 4);         //  8 MB
  __bf16* clo   = (__bf16*)(ws + (size_t)B_N * D_N * 4 + (size_t)K_N * D_N * 2);
  float*  csq   = (float*)(ws + (size_t)B_N * D_N * 4 + (size_t)K_N * D_N * 4);
  float*  cnts  = (float*)((char*)csq + (size_t)K_N * 4);
  int*    idxw  = (int*)((char*)cnts + (size_t)K_N * 4);

  k_convert_z <<<(B_N * (size_t)D_N) / 256, 256, 0, stream>>>(z_e, zhi, zlo);
  k_convert_cb<<<K_N / 8, 256, 0, stream>>>(cb, chi, clo, csq);
  k_init      <<<((size_t)K_N * D_N) / 256, 256, 0, stream>>>(ecb, necb, cnts);
  k_argmin    <<<B_N / 128, 256, 2 * STRIP_BYTES, stream>>>(zhi, zlo, chi, clo,
                                                            csq, idxw, idx_f);
  k_scatter   <<<((size_t)B_N * D_N) / 256, 256, 0, stream>>>(z_e, cb, idxw, zq, necb, cnts);
  k_cluster   <<<K_N / 256, 256, 0, stream>>>(ecs, cnts, ncs);
  k_newcb     <<<((size_t)K_N * D_N) / 256, 256, 0, stream>>>(necb, ecs, cnts, ncb);
}
